// Encoder_50294067036733
// MI455X (gfx1250) — matrix operand order repaired and checked by reading, not compiled, not run on hardware
//
#include <hip/hip_runtime.h>
#include <math.h>

// ---------------------------------------------------------------------------
// BiDAF encoder for MI455X (gfx1250, wave32, WMMA 16x16x32 f16 -> f32 acc)
// ---------------------------------------------------------------------------

typedef __attribute__((ext_vector_type(16))) _Float16 v16h;
typedef __attribute__((ext_vector_type(8)))  float    v8f;

constexpr int E_  = 256;   // embedding dim
constexpr int H_  = 256;   // hidden per direction
constexpr int D_  = 512;   // 2H
constexpr int TP_ = 384;
constexpr int TQ_ = 64;
constexpr int B_  = 32;
constexpr int G3H = 768;   // 3*H

#define WMMA(a,b,c) __builtin_amdgcn_wmma_f32_16x16x32_f16(false,(a),false,(b),(short)0,(c),false,false)

#ifndef __has_builtin
#define __has_builtin(x) 0
#endif
#if __has_builtin(__builtin_amdgcn_global_load_async_to_lds_b128) && \
    __has_builtin(__builtin_amdgcn_s_wait_asynccnt)
#define USE_ASYNC_LDS 1
// The builtin's pointer params are int-vector typed (per hipcc diagnostic):
//   void (v4i addrspace(1)* src, v4i addrspace(3)* dst, imm offset, imm cpol)
typedef int v4i_vs __attribute__((vector_size(16)));
typedef __attribute__((address_space(1))) v4i_vs* async_gptr_t;
typedef __attribute__((address_space(3))) v4i_vs* async_lptr_t;
#else
#define USE_ASYNC_LDS 0
#endif

// ---- WMMA operand loaders (CDNA5 ISA 7.12.2 layouts) ----------------------
// A: 16x32 (MxK) f16.  lane L: M = L%16, khalf = L/16.
// halves i: K = khalf*16 + i  (contiguous, matches B loaders)
__device__ __forceinline__ v16h loadA16(const _Float16* A, int ld, int row0,
                                        int k0, int lane) {
  int m  = lane & 15;
  int kh = (lane >> 4) & 1;
  const _Float16* base = A + (row0 + m) * ld + k0;
  v16h a;
#pragma unroll
  for (int i = 0; i < 16; ++i) {
    int v = i >> 1, j = i & 1;
    int K = kh * 16 + v * 2 + j;
    a[i] = base[K];
  }
  return a;
}

// B: 32x16 (KxN) f16 from W (M x K) row-major used transposed: B[k][n]=W[n0+n][k0+k]
// lane L: N = L%16, kgroup = L/16; halves i: K = kgroup*16 + i  (contiguous!)
__device__ __forceinline__ v16h loadB_T(const _Float16* W, int ld, int n0,
                                        int k0, int lane) {
  int n  = lane & 15;
  int kg = (lane >> 4) & 1;
  const _Float16* src = W + (size_t)(n0 + n) * ld + k0 + kg * 16;
  v16h b;
#pragma unroll
  for (int i = 0; i < 16; ++i) b[i] = src[i];
  return b;
}

// B from row-major (K x N) matrix (not transposed): B[k][n] = M[(k0+k)*ld + n0+n]
__device__ __forceinline__ v16h loadB_N(const _Float16* Mt, int ld, int n0,
                                        int k0, int lane) {
  int n  = lane & 15;
  int kg = (lane >> 4) & 1;
  v16h b;
#pragma unroll
  for (int i = 0; i < 16; ++i) b[i] = Mt[(size_t)(k0 + kg * 16 + i) * ld + n0 + n];
  return b;
}

// ---------------------------------------------------------------------------
// small elementwise kernels
// ---------------------------------------------------------------------------
__global__ void k_f32_to_f16(const float* __restrict__ s, _Float16* __restrict__ d,
                             size_t n) {
  for (size_t i = (size_t)blockIdx.x * blockDim.x + threadIdx.x; i < n;
       i += (size_t)gridDim.x * blockDim.x)
    d[i] = (_Float16)s[i];
}

__global__ void k_embed(const int* __restrict__ idx, const float* __restrict__ emb,
                        _Float16* __restrict__ out, int nrows) {
  size_t i = (size_t)blockIdx.x * blockDim.x + threadIdx.x;
  if (i >= (size_t)nrows * E_) return;
  int row = (int)(i >> 8), d = (int)(i & 255);
  int tok = idx[row];
  out[i] = (_Float16)emb[(size_t)tok * E_ + d];
}

// cast f32->f16 and optionally produce x*wm (wm indexed by col = i % D_)
__global__ void k_cast(const float* __restrict__ x, const float* __restrict__ wm,
                       _Float16* __restrict__ x16, _Float16* __restrict__ xm16,
                       size_t n) {
  size_t i = (size_t)blockIdx.x * blockDim.x + threadIdx.x;
  if (i >= n) return;
  float v = x[i];
  if (x16)  x16[i]  = (_Float16)v;
  if (xm16) xm16[i] = (_Float16)(v * wm[i % D_]);
}

// one wave per row: out[row] = dot(x[row, 0:512], w)
__global__ __launch_bounds__(256) void k_dotw(const float* __restrict__ x,
                                              const float* __restrict__ w,
                                              float* __restrict__ out, int rows) {
  int row  = blockIdx.x * 8 + (threadIdx.x >> 5);
  int lane = threadIdx.x & 31;
  if (row >= rows) return;
  const float* xr = x + (size_t)row * D_;
  float s = 0.f;
  for (int d = lane; d < D_; d += 32) s += xr[d] * w[d];
#pragma unroll
  for (int off = 16; off; off >>= 1) s += __shfl_xor(s, off, 32);
  if (lane == 0) out[row] = s;
}

// ---------------------------------------------------------------------------
// GEMM: C(N x M) = X(N x K, f16 rm) * W(M x K, f16 rm)^T + bias
// Block = 8 waves = macro-tile 32 rows x 256 cols. The 32x32 A stripe is
// shared by all waves: staged to LDS with gfx1250 async-to-LDS (ASYNCcnt)
// double buffering. Each wave computes a 32x32 output (2x2 WMMA register
// block -> 4 WMMAs per 2 A-loads + 2 B-loads).
// Grid must be exactly (N/32)*(M/256) blocks.
// ---------------------------------------------------------------------------
__global__ __launch_bounds__(256) void k_gemm_xwT(const _Float16* __restrict__ X,
                                                  const _Float16* __restrict__ W,
                                                  const float* __restrict__ bias,
                                                  float* __restrict__ C,
                                                  int N, int M, int K) {
  __shared__ __align__(16) _Float16 As[2][32 * 32];  // 2 x 2KB double buffer
  int tid  = threadIdx.x;
  int wave = tid >> 5, lane = tid & 31;

  int mt = M >> 8;                    // macro-cols of 256
  int bi = blockIdx.x;
  int r0   = (bi / mt) << 5;          // 32-row stripe
  int c0   = ((bi % mt) << 8) + (wave << 5);  // this wave's 32 columns

  auto stage = [&](int buf, int kk) {
#if USE_ASYNC_LDS
    if (wave == 0) {
#pragma unroll
      for (int rep = 0; rep < 4; ++rep) {
        int idx = rep * 32 + lane;            // 128 x 16B = 2KB tile
        int row = idx >> 2, seg = idx & 3;
        const _Float16* g = X + (size_t)(r0 + row) * K + kk + seg * 8;
        __builtin_amdgcn_global_load_async_to_lds_b128(
            (async_gptr_t)g, (async_lptr_t)&As[buf][idx * 8], 0, 0);
      }
    }
#else
    int row = tid >> 3, seg = tid & 7;        // 256 threads x 8B
    const _Float16* g = X + (size_t)(r0 + row) * K + kk + seg * 4;
    _Float16* d = &As[buf][row * 32 + seg * 4];
#pragma unroll
    for (int j = 0; j < 4; ++j) d[j] = g[j];
#endif
  };

  v8f acc00 = {}, acc01 = {}, acc10 = {}, acc11 = {};
  stage(0, 0);
  for (int kk = 0; kk < K; kk += 32) {
    int cur = (kk >> 5) & 1;
#if USE_ASYNC_LDS
    __builtin_amdgcn_s_wait_asynccnt(0);      // buf[cur] landed in LDS
#endif
    __syncthreads();                          // publish buf[cur] to all waves
    if (kk + 32 < K) stage(cur ^ 1, kk + 32); // overlap next copy with compute
    v16h a0 = loadA16(As[cur], 32, 0,  0, lane);
    v16h a1 = loadA16(As[cur], 32, 16, 0, lane);
    v16h b0 = loadB_T(W, K, c0,      kk, lane);
    v16h b1 = loadB_T(W, K, c0 + 16, kk, lane);
    acc00 = WMMA(a0, b0, acc00);
    acc01 = WMMA(a0, b1, acc01);
    acc10 = WMMA(a1, b0, acc10);
    acc11 = WMMA(a1, b1, acc11);
    __syncthreads();                          // reads done before re-staging
  }
  int n = lane & 15, mb = (lane >> 4) * 8;
  float bv0 = bias ? bias[c0 + n] : 0.f;
  float bv1 = bias ? bias[c0 + 16 + n] : 0.f;
#pragma unroll
  for (int r = 0; r < 8; ++r) {
    size_t ro0 = (size_t)(r0 + mb + r) * M;
    size_t ro1 = (size_t)(r0 + 16 + mb + r) * M;
    C[ro0 + c0 + n]      = acc00[r] + bv0;
    C[ro0 + c0 + 16 + n] = acc01[r] + bv1;
    C[ro1 + c0 + n]      = acc10[r] + bv0;
    C[ro1 + c0 + 16 + n] = acc11[r] + bv1;
  }
}

// ---------------------------------------------------------------------------
// persistent bidirectional GRU step kernel.
// grid.x = 2 (dir 0 fwd / 1 bwd), block = 1024 (32 waves).
// wave w: hidden cols [(w/2)*16 ..), batch rows [(w%2)*16 ..).
// Each wave computes its r,z,n gate tiles with 3 WMMAs per k-step; gate math
// stays in the accumulator registers -> no gh staging.
// ---------------------------------------------------------------------------
__global__ __launch_bounds__(1024) void k_gru(
    const float* __restrict__ gi_f, const float* __restrict__ gi_b,  // (T*B,3H)
    const _Float16* __restrict__ whh_f, const _Float16* __restrict__ whh_b,
    const float* __restrict__ bhh_f, const float* __restrict__ bhh_b,
    float* __restrict__ y,  // (T,B,2H); fwd -> cols [0,H), bwd -> [H,2H)
    int T) {
  int dir = blockIdx.x;
  const float*    gi  = dir ? gi_b  : gi_f;
  const _Float16* whh = dir ? whh_b : whh_f;
  const float*    bhh = dir ? bhh_b : bhh_f;
  int colbase = dir ? H_ : 0;

  __shared__ __align__(16) _Float16 hsh[B_ * H_];  // 16 KB f16 state

  int tid  = threadIdx.x;
  int wave = tid >> 5, lane = tid & 31;
  int j0   = (wave >> 1) << 4;  // hidden column base
  int row0 = (wave & 1) << 4;   // batch row base

  for (int i = tid; i < B_ * H_; i += blockDim.x) hsh[i] = (_Float16)0.f;
  __syncthreads();

  int ncol  = lane & 15;
  int mbase = (lane >> 4) * 8;
  float b_r = bhh[j0 + ncol];
  float b_z = bhh[H_ + j0 + ncol];
  float b_n = bhh[2 * H_ + j0 + ncol];

  for (int s = 0; s < T; ++s) {
    int t = dir ? (T - 1 - s) : s;
    v8f cr = {}, cz = {}, cn = {};
#pragma unroll
    for (int kk = 0; kk < H_; kk += 32) {
      v16h a  = loadA16(hsh, H_, row0, kk, lane);
      v16h br = loadB_T(whh, H_, j0,          kk, lane);
      v16h bz = loadB_T(whh, H_, H_ + j0,     kk, lane);
      v16h bn = loadB_T(whh, H_, 2 * H_ + j0, kk, lane);
      cr = WMMA(a, br, cr);
      cz = WMMA(a, bz, cz);
      cn = WMMA(a, bn, cn);
    }
    float hnew[8];
#pragma unroll
    for (int r = 0; r < 8; ++r) {
      int brow = row0 + mbase + r;
      int col  = j0 + ncol;
      const float* gr = gi + ((size_t)t * B_ + brow) * G3H;
      float ir = gr[col], iz = gr[H_ + col], inn = gr[2 * H_ + col];
      float hold = (float)hsh[brow * H_ + col];
      float rg = 1.f / (1.f + __expf(-(ir + cr[r] + b_r)));
      float zg = 1.f / (1.f + __expf(-(iz + cz[r] + b_z)));
      float ng = tanhf(inn + rg * (cn[r] + b_n));
      hnew[r] = (1.f - zg) * ng + zg * hold;
    }
    __syncthreads();  // all reads of hsh done
#pragma unroll
    for (int r = 0; r < 8; ++r) {
      int brow = row0 + mbase + r;
      int col  = j0 + ncol;
      hsh[brow * H_ + col] = (_Float16)hnew[r];
      y[((size_t)t * B_ + brow) * D_ + colbase + col] = hnew[r];
    }
    __syncthreads();  // new h visible before next step
  }
}

// ---------------------------------------------------------------------------
// BiDAF similarity: sim[p,q,b] = sp[p,b] + sq[q,b] + <pm[p,b,:], q[q,b,:]>
// per-batch WMMA GEMM; one wave per 16x16 (p,q) tile.
// ---------------------------------------------------------------------------
__global__ __launch_bounds__(256) void k_sim(const _Float16* __restrict__ pm,
                                             const _Float16* __restrict__ q16,
                                             const float* __restrict__ sp,
                                             const float* __restrict__ sq,
                                             float* __restrict__ sim) {
  const int tilesP = TP_ / 16, tilesQ = TQ_ / 16;
  int wid = blockIdx.x * 8 + (threadIdx.x >> 5);
  int lane = threadIdx.x & 31;
  int b = wid / (tilesP * tilesQ);
  if (b >= B_) return;
  int rem = wid % (tilesP * tilesQ);
  int p0 = (rem / tilesQ) << 4, q0 = (rem % tilesQ) << 4;
  const _Float16* pb = pm  + b * D_;   // ld = B*D
  const _Float16* qb = q16 + b * D_;
  v8f acc = {};
  for (int kk = 0; kk < D_; kk += 32) {
    v16h a  = loadA16(pb, B_ * D_, p0, kk, lane);
    v16h bm = loadB_T(qb, B_ * D_, q0, kk, lane);
    acc = WMMA(a, bm, acc);
  }
  int n = lane & 15, mb = (lane >> 4) * 8;
#pragma unroll
  for (int r = 0; r < 8; ++r) {
    int p = p0 + mb + r, q = q0 + n;
    sim[((size_t)p * TQ_ + q) * B_ + b] = acc[r] + sp[p * B_ + b] + sq[q * B_ + b];
  }
}

// softmax over q (Tq=64) per (p,b) row; also emit pre-softmax row max.
// c2q written as (b, Tp, Tq) f16 for the next WMMA A-operand.
__global__ __launch_bounds__(256) void k_softmax_q(const float* __restrict__ sim,
                                                   _Float16* __restrict__ c2q,
                                                   float* __restrict__ rowmax) {
  int row = blockIdx.x * 8 + (threadIdx.x >> 5);
  int lane = threadIdx.x & 31;
  if (row >= TP_ * B_) return;
  int p = row / B_, b = row % B_;
  const float* s = sim + (size_t)p * TQ_ * B_ + b;
  float v0 = s[(size_t)lane * B_], v1 = s[(size_t)(lane + 32) * B_];
  float mx = fmaxf(v0, v1);
#pragma unroll
  for (int off = 16; off; off >>= 1) mx = fmaxf(mx, __shfl_xor(mx, off, 32));
  float e0 = __expf(v0 - mx), e1 = __expf(v1 - mx);
  float sm = e0 + e1;
#pragma unroll
  for (int off = 16; off; off >>= 1) sm += __shfl_xor(sm, off, 32);
  float inv = 1.f / sm;
  _Float16* dst = c2q + ((size_t)b * TP_ + p) * TQ_;
  dst[lane]      = (_Float16)(e0 * inv);
  dst[lane + 32] = (_Float16)(e1 * inv);
  if (lane == 0) rowmax[p * B_ + b] = mx;
}

// c2q_vec[p,b,:] = c2q[b,p,:] @ q_out[:,b,:]   (per-batch Tp x 64 x 512 GEMM)
__global__ __launch_bounds__(256) void k_c2qvec(const _Float16* __restrict__ c2q,
                                                const _Float16* __restrict__ q16,
                                                float* __restrict__ c2qv) {
  const int tilesP = TP_ / 16, tilesD = D_ / 16;
  int wid = blockIdx.x * 8 + (threadIdx.x >> 5);
  int lane = threadIdx.x & 31;
  int b = wid / (tilesP * tilesD);
  if (b >= B_) return;
  int rem = wid % (tilesP * tilesD);
  int p0 = (rem / tilesD) << 4, d0 = (rem % tilesD) << 4;
  const _Float16* A = c2q + (size_t)b * TP_ * TQ_;
  const _Float16* Q = q16 + b * D_;
  v8f acc = {};
  for (int kk = 0; kk < TQ_; kk += 32) {
    v16h a  = loadA16(A, TQ_, p0, kk, lane);
    v16h bm = loadB_N(Q, B_ * D_, d0, kk, lane);
    acc = WMMA(a, bm, acc);
  }
  int n = lane & 15, mb = (lane >> 4) * 8;
#pragma unroll
  for (int r = 0; r < 8; ++r)
    c2qv[((size_t)(p0 + mb + r) * B_ + b) * D_ + d0 + n] = acc[r];
}

// q2c: att = softmax_p(rowmax[:,b]); q2c_vec[b,:] = sum_p att[p]*p_out[p,b,:]
__global__ __launch_bounds__(256) void k_q2c(const float* __restrict__ rowmax,
                                             const float* __restrict__ pout,
                                             float* __restrict__ q2cv) {
  int b = blockIdx.x, tid = threadIdx.x;
  __shared__ float att[TP_];
  __shared__ float red[256];
  float mx = -3.4e38f;
  for (int p = tid; p < TP_; p += 256) mx = fmaxf(mx, rowmax[p * B_ + b]);
  red[tid] = mx; __syncthreads();
  for (int s = 128; s; s >>= 1) { if (tid < s) red[tid] = fmaxf(red[tid], red[tid + s]); __syncthreads(); }
  mx = red[0]; __syncthreads();
  float sm = 0.f;
  for (int p = tid; p < TP_; p += 256) { float e = __expf(rowmax[p * B_ + b] - mx); att[p] = e; sm += e; }
  red[tid] = sm; __syncthreads();
  for (int s = 128; s; s >>= 1) { if (tid < s) red[tid] += red[tid + s]; __syncthreads(); }
  float inv = 1.f / red[0];
  __syncthreads();
  for (int d = tid; d < D_; d += 256) {
    float acc = 0.f;
    for (int p = 0; p < TP_; ++p) acc += att[p] * pout[((size_t)p * B_ + b) * D_ + d];
    q2cv[b * D_ + d] = acc * inv;
  }
}

// G[row, :] = [p_out | c2q_vec | p_out*c2q_vec | p_out*q2c_vec]  (f16, 2048 wide)
__global__ void k_buildG(const float* __restrict__ pout, const float* __restrict__ c2qv,
                         const float* __restrict__ q2cv, _Float16* __restrict__ G) {
  size_t i = (size_t)blockIdx.x * blockDim.x + threadIdx.x;
  if (i >= (size_t)TP_ * B_ * D_) return;
  size_t row = i / D_; int d = (int)(i % D_);
  int b = (int)(row % B_);
  float po = pout[i], cv = c2qv[i], qv = q2cv[b * D_ + d];
  _Float16* g = G + row * (4 * D_);
  g[d]          = (_Float16)po;
  g[D_ + d]     = (_Float16)cv;
  g[2 * D_ + d] = (_Float16)(po * cv);
  g[3 * D_ + d] = (_Float16)(po * qv);
}

// p_hn: [0] = fwd final h = y[T-1,:, :H]; [1] = bwd final h = y[0,:, H:2H]
__global__ void k_phn(const float* __restrict__ y, float* __restrict__ hn) {
  int i = blockIdx.x * blockDim.x + threadIdx.x;
  if (i >= 2 * B_ * H_) return;
  int dir = i / (B_ * H_), b = (i / H_) % B_, j = i % H_;
  hn[i] = dir == 0 ? y[((size_t)(TP_ - 1) * B_ + b) * D_ + j]
                   : y[((size_t)b) * D_ + H_ + j];
}

// ---------------------------------------------------------------------------
extern "C" void kernel_launch(void* const* d_in, const int* in_sizes, int n_in,
                              void* d_out, int out_size, void* d_ws, size_t ws_size,
                              hipStream_t stream) {
  const int*   p_batch = (const int*)d_in[0];
  const int*   q_batch = (const int*)d_in[1];
  const float* emb     = (const float*)d_in[2];
  const float* w1_ih_f = (const float*)d_in[3];
  const float* w1_hh_f = (const float*)d_in[4];
  const float* b1_ih_f = (const float*)d_in[5];
  const float* b1_hh_f = (const float*)d_in[6];
  const float* w1_ih_b = (const float*)d_in[7];
  const float* w1_hh_b = (const float*)d_in[8];
  const float* b1_ih_b = (const float*)d_in[9];
  const float* b1_hh_b = (const float*)d_in[10];
  const float* bidaf_w = (const float*)d_in[11];
  const float* w4_ih_f = (const float*)d_in[12];
  const float* w4_hh_f = (const float*)d_in[13];
  const float* b4_ih_f = (const float*)d_in[14];
  const float* b4_hh_f = (const float*)d_in[15];
  const float* w4_ih_b = (const float*)d_in[16];
  const float* w4_hh_b = (const float*)d_in[17];
  const float* b4_ih_b = (const float*)d_in[18];
  const float* b4_hh_b = (const float*)d_in[19];

  // bump allocator over d_ws
  char* ws = (char*)d_ws;
  size_t off = 0;
  auto alloc = [&](size_t bytes) -> void* {
    off = (off + 255) & ~(size_t)255;
    void* p = ws + off;
    off += bytes;
    return p;
  };
  const int NP = TP_ * B_;  // 12288
  const int NQ = TQ_ * B_;  // 2048

  _Float16* w1ih16_f = (_Float16*)alloc((size_t)G3H * E_ * 2);
  _Float16* w1ih16_b = (_Float16*)alloc((size_t)G3H * E_ * 2);
  _Float16* w1hh16_f = (_Float16*)alloc((size_t)G3H * H_ * 2);
  _Float16* w1hh16_b = (_Float16*)alloc((size_t)G3H * H_ * 2);
  _Float16* w4ih16_f = (_Float16*)alloc((size_t)G3H * 4 * D_ * 2);
  _Float16* w4ih16_b = (_Float16*)alloc((size_t)G3H * 4 * D_ * 2);
  _Float16* w4hh16_f = (_Float16*)alloc((size_t)G3H * H_ * 2);
  _Float16* w4hh16_b = (_Float16*)alloc((size_t)G3H * H_ * 2);
  _Float16* pemb16   = (_Float16*)alloc((size_t)NP * E_ * 2);
  _Float16* qemb16   = (_Float16*)alloc((size_t)NQ * E_ * 2);
  float* gi_p_f = (float*)alloc((size_t)NP * G3H * 4);
  float* gi_p_b = (float*)alloc((size_t)NP * G3H * 4);
  float* gi_q_f = (float*)alloc((size_t)NQ * G3H * 4);
  float* gi_q_b = (float*)alloc((size_t)NQ * G3H * 4);
  float* p_out  = (float*)alloc((size_t)NP * D_ * 4);
  float* q_out  = (float*)alloc((size_t)NQ * D_ * 4);
  _Float16* qf16 = (_Float16*)alloc((size_t)NQ * D_ * 2);
  _Float16* pm16 = (_Float16*)alloc((size_t)NP * D_ * 2);
  float* sp     = (float*)alloc((size_t)NP * 4);
  float* sq     = (float*)alloc((size_t)NQ * 4);
  float* sim    = (float*)alloc((size_t)TP_ * TQ_ * B_ * 4);
  float* rowmax = (float*)alloc((size_t)NP * 4);
  _Float16* c2q16 = (_Float16*)alloc((size_t)B_ * TP_ * TQ_ * 2);
  float* c2qv   = (float*)alloc((size_t)NP * D_ * 4);
  float* q2cv   = (float*)alloc((size_t)B_ * D_ * 4);
  _Float16* G16 = (_Float16*)alloc((size_t)NP * 4 * D_ * 2);
  float* gi4_f  = gi_p_f;  // reuse: rnn1-p gi dead by the time rnn4 runs
  float* gi4_b  = gi_p_b;

  float* y_out = (float*)d_out;                  // (Tp,B,2H)
  float* hn    = y_out + (size_t)TP_ * B_ * D_;  // (2,B,H)

  // 1) weights -> f16
  struct { const float* s; _Float16* d; size_t n; } cv[8] = {
    {w1_ih_f, w1ih16_f, (size_t)G3H * E_}, {w1_ih_b, w1ih16_b, (size_t)G3H * E_},
    {w1_hh_f, w1hh16_f, (size_t)G3H * H_}, {w1_hh_b, w1hh16_b, (size_t)G3H * H_},
    {w4_ih_f, w4ih16_f, (size_t)G3H * 4 * D_}, {w4_ih_b, w4ih16_b, (size_t)G3H * 4 * D_},
    {w4_hh_f, w4hh16_f, (size_t)G3H * H_}, {w4_hh_b, w4hh16_b, (size_t)G3H * H_}};
  for (int i = 0; i < 8; ++i) {
    int blocks = (int)((cv[i].n + 255) / 256); if (blocks > 4096) blocks = 4096;
    k_f32_to_f16<<<blocks, 256, 0, stream>>>(cv[i].s, cv[i].d, cv[i].n);
  }

  // 2) embedding gather -> f16
  k_embed<<<(NP * E_ + 255) / 256, 256, 0, stream>>>(p_batch, emb, pemb16, NP);
  k_embed<<<(NQ * E_ + 255) / 256, 256, 0, stream>>>(q_batch, emb, qemb16, NQ);

  // 3) rnn1 input projections (batched over time) + bias
  k_gemm_xwT<<<(NP / 32) * (G3H / 256), 256, 0, stream>>>(pemb16, w1ih16_f, b1_ih_f, gi_p_f, NP, G3H, E_);
  k_gemm_xwT<<<(NP / 32) * (G3H / 256), 256, 0, stream>>>(pemb16, w1ih16_b, b1_ih_b, gi_p_b, NP, G3H, E_);
  k_gemm_xwT<<<(NQ / 32) * (G3H / 256), 256, 0, stream>>>(qemb16, w1ih16_f, b1_ih_f, gi_q_f, NQ, G3H, E_);
  k_gemm_xwT<<<(NQ / 32) * (G3H / 256), 256, 0, stream>>>(qemb16, w1ih16_b, b1_ih_b, gi_q_b, NQ, G3H, E_);

  // 4) rnn1 recurrence (both directions concurrently per launch)
  k_gru<<<2, 1024, 0, stream>>>(gi_p_f, gi_p_b, w1hh16_f, w1hh16_b, b1_hh_f, b1_hh_b, p_out, TP_);
  k_gru<<<2, 1024, 0, stream>>>(gi_q_f, gi_q_b, w1hh16_f, w1hh16_b, b1_hh_f, b1_hh_b, q_out, TQ_);

  // 5) attention operands
  k_cast<<<(int)(((size_t)NP * D_ + 255) / 256), 256, 0, stream>>>(p_out, bidaf_w + 2 * D_, (_Float16*)nullptr, pm16, (size_t)NP * D_);
  k_cast<<<(int)(((size_t)NQ * D_ + 255) / 256), 256, 0, stream>>>(q_out, bidaf_w, qf16, (_Float16*)nullptr, (size_t)NQ * D_);
  k_dotw<<<(NP + 7) / 8, 256, 0, stream>>>(p_out, bidaf_w, sp, NP);
  k_dotw<<<(NQ + 7) / 8, 256, 0, stream>>>(q_out, bidaf_w + D_, sq, NQ);

  // 6) similarity + softmaxes + attention vectors
  k_sim<<<B_ * (TP_ / 16) * (TQ_ / 16) / 8, 256, 0, stream>>>(pm16, qf16, sp, sq, sim);
  k_softmax_q<<<(NP + 7) / 8, 256, 0, stream>>>(sim, c2q16, rowmax);
  k_c2qvec<<<B_ * (TP_ / 16) * (D_ / 16) / 8, 256, 0, stream>>>(c2q16, qf16, c2qv);
  k_q2c<<<B_, 256, 0, stream>>>(rowmax, p_out, q2cv);

  // 7) G assembly
  k_buildG<<<(int)(((size_t)NP * D_ + 255) / 256), 256, 0, stream>>>(p_out, c2qv, q2cv, G16);

  // 8) rnn4 input projections (K=2048: the dominant GEMMs)
  k_gemm_xwT<<<(NP / 32) * (G3H / 256), 256, 0, stream>>>(G16, w4ih16_f, b4_ih_f, gi4_f, NP, G3H, 4 * D_);
  k_gemm_xwT<<<(NP / 32) * (G3H / 256), 256, 0, stream>>>(G16, w4ih16_b, b4_ih_b, gi4_b, NP, G3H, 4 * D_);

  // 9) rnn4 recurrence writes p_outputs directly into d_out
  k_gru<<<2, 1024, 0, stream>>>(gi4_f, gi4_b, w4hh16_f, w4hh16_b, b4_hh_f, b4_hh_b, y_out, TP_);

  // 10) final hidden states
  k_phn<<<(2 * B_ * H_ + 255) / 256, 256, 0, stream>>>(y_out, hn);

  (void)in_sizes; (void)n_in; (void)out_size; (void)ws_size;
}
